// WindowAttention_20830591386187
// MI455X (gfx1250) — compile-verified, matrix operand order checked by reading, not run on hardware
//
#include <hip/hip_runtime.h>
#include <hip/hip_bf16.h>

#define LQ   4096
#define NHD  16
#define HD   64
#define QCH  128     // queries per block
#define KWIN 256     // padded key window (254 -> 256)
#define WRAD 63
#define VTS  266     // Vt row stride (keys, padded: 133 dwords, coprime with 64 banks)

typedef __attribute__((ext_vector_type(16))) __bf16 v16bf;
typedef __attribute__((ext_vector_type(8)))  float  v8f;

union Frag {
  v16bf v;
  unsigned short u[16];
  unsigned int   d[8];
};

__device__ __forceinline__ unsigned short f2bf(float f) {
  unsigned int u = __float_as_uint(f);
  u += 0x7FFFu + ((u >> 16) & 1u);        // round-to-nearest-even
  return (unsigned short)(u >> 16);
}

__device__ __forceinline__ float fast_exp2(float x) {
  return __builtin_amdgcn_exp2f(x);       // v_exp_f32 (native base-2)
}

__global__ __launch_bounds__(256, 1)
void WindowAttention_wmma_kernel(const float* __restrict__ qp,
                                 const float* __restrict__ kp,
                                 const float* __restrict__ vp,
                                 float* __restrict__ out) {
  __shared__ unsigned short Qs[QCH * 72];    // bf16 Q[q][d], padded stride 72
  __shared__ unsigned short Ks[KWIN * 72];   // bf16 K[key][d], padded stride 72
  __shared__ unsigned short Vt[HD * VTS];    // bf16 V^T[d][key], stride 266
  __shared__ float          Os[QCH * 68];    // f32 O[q][d], padded stride 68

  const int tid  = threadIdx.x;
  const int lane = tid & 31;
  const int wave = tid >> 5;
  const int ln   = lane & 15;
  const int g    = lane >> 4;

  const int blk = blockIdx.x;
  const int c   = blk & 31;            // 32 query chunks (fastest -> L2 locality)
  const int nh  = (blk >> 5) & 15;     // 16 heads
  const int b   = blk >> 9;            // 4 batches

  const int qbase = c * QCH;
  const int kbase = qbase - WRAD;

  // ---------------- cooperative staging: global fp32 -> LDS bf16 ----------------
  // Q: fold h^-0.5 and log2(e) into the conversion (softmax done in base 2)
  const float QSCALE = 0.125f * 1.44269504088896340736f;
  for (int i = 0; i < 16; ++i) {
    const int row = wave * 16 + i;
    const float* src = qp + (((size_t)(b * LQ + qbase + row) * NHD + nh) * HD);
    float2 x = *(const float2*)(src + lane * 2);
    Qs[row * 72 + lane * 2 + 0] = f2bf(x.x * QSCALE);
    Qs[row * 72 + lane * 2 + 1] = f2bf(x.y * QSCALE);
  }
  // K row-major; V transposed (Vt[d][key]); zero-pad outside the sequence
  for (int i = 0; i < 32; ++i) {
    const int row = wave * 32 + i;
    const int kg  = kbase + row;
    if (kg >= 0 && kg < LQ) {
      const size_t base = ((size_t)(b * LQ + kg) * NHD + nh) * HD;
      float2 xk = *(const float2*)(kp + base + lane * 2);
      float2 xv = *(const float2*)(vp + base + lane * 2);
      Ks[row * 72 + lane * 2 + 0] = f2bf(xk.x);
      Ks[row * 72 + lane * 2 + 1] = f2bf(xk.y);
      Vt[(lane * 2 + 0) * VTS + row] = f2bf(xv.x);
      Vt[(lane * 2 + 1) * VTS + row] = f2bf(xv.y);
    } else {
      Ks[row * 72 + lane * 2 + 0] = 0; Ks[row * 72 + lane * 2 + 1] = 0;
      Vt[(lane * 2 + 0) * VTS + row] = 0;
      Vt[(lane * 2 + 1) * VTS + row] = 0;
    }
  }
  __syncthreads();

  // ---------------- per-wave: 16-query strip, flash-style over 8 key chunks -----
  // Q^T as B-fragment (column q = ln, element -> head dim), two K=32 halves
  Frag qb[2];
  const int qrow = wave * 16 + ln;
  #pragma unroll
  for (int half = 0; half < 2; ++half)
    #pragma unroll
    for (int e2 = 0; e2 < 8; ++e2) {
      const int off = half * 32 + 2 * (e2 & 3) + (g << 3) + ((e2 & 4) << 2);
      qb[half].d[e2] = *(const unsigned int*)&Qs[qrow * 72 + off];
    }

  v8f zero8 = {0.f,0.f,0.f,0.f,0.f,0.f,0.f,0.f};
  v8f oacc[4];
  #pragma unroll
  for (int nt = 0; nt < 4; ++nt) oacc[nt] = zero8;
  float mrun = -1.0e30f, lrun = 0.f;   // log2-domain running max / sum
  const int qloc = wave * 16 + ln;

  for (int kc = 0; kc < 8; ++kc) {
    // S^T = K_tile x Q^T  (two 16-key tiles, head dim 64 = 2 x K32)
    v8f s0 = zero8, s1 = zero8;
    #pragma unroll
    for (int half = 0; half < 2; ++half) {
      Frag ka, kb;
      #pragma unroll
      for (int e2 = 0; e2 < 8; ++e2) {
        const int off = half * 32 + 2 * (e2 & 3) + (g << 3) + ((e2 & 4) << 2);
        ka.d[e2] = *(const unsigned int*)&Ks[(kc * 32 +      ln) * 72 + off];
        kb.d[e2] = *(const unsigned int*)&Ks[(kc * 32 + 16 + ln) * 72 + off];
      }
      s0 = __builtin_amdgcn_wmma_f32_16x16x32_bf16(false, ka.v, false, qb[half].v,
                                                   (short)0, s0, false, false);
      s1 = __builtin_amdgcn_wmma_f32_16x16x32_bf16(false, kb.v, false, qb[half].v,
                                                   (short)0, s1, false, false);
    }

    // band + sequence-edge mask, then online softmax (query lives on lane%16)
    float mc = -3.0e38f;
    #pragma unroll
    for (int vv = 0; vv < 8; ++vv) {
      const int kl0 = kc * 32 + vv + (g << 3);
      const int kl1 = kl0 + 16;
      const int d0 = kl0 - qloc, d1 = kl1 - qloc;
      const int kg0 = kbase + kl0, kg1 = kbase + kl1;
      const bool ok0 = ((unsigned)d0 <= 126u) & ((unsigned)kg0 < (unsigned)LQ);
      const bool ok1 = ((unsigned)d1 <= 126u) & ((unsigned)kg1 < (unsigned)LQ);
      s0[vv] = ok0 ? s0[vv] : -3.0e38f;
      s1[vv] = ok1 ? s1[vv] : -3.0e38f;
      mc = fmaxf(mc, fmaxf(s0[vv], s1[vv]));
    }
    mc = fmaxf(mc, __shfl_xor(mc, 16, 32));
    const float mnew = fmaxf(mrun, mc);
    const float corr = fast_exp2(mrun - mnew);

    // exp2 + pack P^T straight into the B-fragment of the AV matmul
    Frag pb;
    float psum = 0.f;
    #pragma unroll
    for (int vv = 0; vv < 8; ++vv) {
      const float p0 = fast_exp2(s0[vv] - mnew);
      const float p1 = fast_exp2(s1[vv] - mnew);
      psum += p0 + p1;
      pb.u[vv]     = f2bf(p0);
      pb.u[vv + 8] = f2bf(p1);
    }
    psum += __shfl_xor(psum, 16, 32);
    lrun = lrun * corr + psum;
    mrun = mnew;

    #pragma unroll
    for (int nt = 0; nt < 4; ++nt)
      #pragma unroll
      for (int j = 0; j < 8; ++j) oacc[nt][j] *= corr;

    // O^T += V^T x P^T   (4 head-dim tiles of 16); Vt rows are key-contiguous
    #pragma unroll
    for (int nt = 0; nt < 4; ++nt) {
      Frag va;
      #pragma unroll
      for (int e2 = 0; e2 < 8; ++e2) {
        const int koffp = kc * 32 + 2 * (e2 & 3) + (g << 3) + ((e2 & 4) << 2);
        va.d[e2] = *(const unsigned int*)&Vt[(nt * 16 + ln) * VTS + koffp];
      }
      oacc[nt] = __builtin_amdgcn_wmma_f32_16x16x32_bf16(false, va.v, false, pb.v,
                                                         (short)0, oacc[nt], false, false);
    }
  }

  // ---------------- normalize, stage O^T -> LDS as O[q][d], coalesced store ------
  const float linv = 1.f / lrun;
  #pragma unroll
  for (int nt = 0; nt < 4; ++nt)
    #pragma unroll
    for (int vv = 0; vv < 8; ++vv) {
      const int dd = nt * 16 + vv + (g << 3);
      Os[(wave * 16 + ln) * 68 + dd] = oacc[nt][vv] * linv;
    }
  __syncthreads();

  for (int i = 0; i < 16; ++i) {
    const int row = wave * 16 + i;
    float2 x = *(const float2*)&Os[row * 68 + lane * 2];
    float* dst = out + (((size_t)(b * LQ + qbase + row) * NHD + nh) * HD);
    *(float2*)(dst + lane * 2) = x;
  }
}

extern "C" void kernel_launch(void* const* d_in, const int* in_sizes, int n_in,
                              void* d_out, int out_size, void* d_ws, size_t ws_size,
                              hipStream_t stream) {
  const float* q = (const float*)d_in[0];
  const float* k = (const float*)d_in[1];
  const float* v = (const float*)d_in[2];
  float* out = (float*)d_out;
  (void)in_sizes; (void)n_in; (void)out_size; (void)d_ws; (void)ws_size;
  // grid: 4 batches * 16 heads * 32 query chunks = 2048 blocks
  WindowAttention_wmma_kernel<<<dim3(2048), dim3(256), 0, stream>>>(q, k, v, out);
}
